// CAModel_10479720202644
// MI455X (gfx1250) — matrix operand order repaired and checked by reading, not compiled, not run on hardware
//
#include <hip/hip_runtime.h>

#define Bn 32
#define Hn 128
#define Wn 128
#define Cn 16
#define HIDn 128

typedef __attribute__((ext_vector_type(8)))  _Float16 v8h;
typedef __attribute__((ext_vector_type(16))) _Float16 v16h;
typedef __attribute__((ext_vector_type(8)))  float    v8f;

struct __align__(16) f4 { float v[4]; };

// A-fragment style load: halves [0..7] and [16..23] relative to p
__device__ __forceinline__ v16h frag_stride16(const _Float16* p) {
  v8h a = *(const v8h*)p;
  v8h b = *(const v8h*)(p + 16);
  v16h r;
#pragma unroll
  for (int i = 0; i < 8; ++i) { r[i] = a[i]; r[i + 8] = b[i]; }
  return r;
}
// B-fragment style load: 16 contiguous halves
__device__ __forceinline__ v16h frag_contig16(const _Float16* p) {
  v8h a = *(const v8h*)p;
  v8h b = *(const v8h*)(p + 8);
  v16h r;
#pragma unroll
  for (int i = 0; i < 8; ++i) { r[i] = a[i]; r[i + 8] = b[i]; }
  return r;
}

// Fused: perception (depthwise sobel) -> relu(y@w0^T) -> h@w1^T -> residual
// update. Each wave32 handles 16 pixels (M=16 of the WMMA tiles); block = 4
// waves = 64 pixels of one image row.
__global__ __launch_bounds__(128) void nca_step_kernel(
    const float* __restrict__ x, const float* __restrict__ w0,
    const float* __restrict__ w1, const float* __restrict__ rmask,
    float* __restrict__ out, float* __restrict__ alpha2) {
  __shared__ __align__(16) _Float16 w0h[128 * 64];     // B for GEMM1: [n][k], K padded 48->64
  __shared__ __align__(16) _Float16 w1t[16 * 128];     // B for GEMM2: [c][k]
  __shared__ __align__(16) _Float16 y_lds[4][16 * 64]; // per-wave A for GEMM1: [m][k]
  __shared__ __align__(16) _Float16 h_lds[4][16 * 128];// per-wave A for GEMM2: [m][k]

  const int tid = threadIdx.x;
  // Stage weights f32 -> f16 (row-major already matches the frag_stride16
  // B-fragment reads; zero-pad K 48..63 for GEMM1).
  for (int i = tid; i < 128 * 64; i += 128) {
    int n = i >> 6, k = i & 63;
    w0h[i] = (k < 48) ? (_Float16)w0[n * 48 + k] : (_Float16)0.0f;
  }
  for (int i = tid; i < 16 * 128; i += 128) w1t[i] = (_Float16)w1[i];

  const int wave = tid >> 5;
  const int lane = tid & 31;
  const int m    = lane & 15;      // pixel (A row) / B column / out channel
  const bool lo  = lane < 16;
  const int hi8  = lo ? 0 : 8;

  const int bh    = blockIdx.x >> 1;
  const int b     = bh / Hn;
  const int hrow  = bh % Hn;
  const int wbase = (blockIdx.x & 1) * 64 + wave * 16;
  const int wcol  = wbase + m;

  // ---- perception: identity + sobel dx/dy, SAME zero padding ----
  // lane<16 computes features 0..23 (ch 0..7), lane>=16 features 24..47.
  const int c0 = lo ? 0 : 8;
  _Float16* yrow = &y_lds[wave][m * 64];
#pragma unroll
  for (int cc = 0; cc < 8; cc += 4) {
    f4 nb[3][3];
#pragma unroll
    for (int i = 0; i < 3; ++i)
#pragma unroll
      for (int j = 0; j < 3; ++j) {
        int rr = hrow + i - 1, qq = wcol + j - 1;
        f4 z; z.v[0] = z.v[1] = z.v[2] = z.v[3] = 0.0f;
        if (rr >= 0 && rr < Hn && qq >= 0 && qq < Wn)
          z = *(const f4*)(x + (((size_t)b * Hn + rr) * Wn + qq) * Cn + c0 + cc);
        nb[i][j] = z;
      }
    _Float16* dst = yrow + (lo ? 0 : 24) + cc * 3;
#pragma unroll
    for (int k = 0; k < 4; ++k) {
      float id  = nb[1][1].v[k];
      float dxv = (nb[0][2].v[k] + 2.f * nb[1][2].v[k] + nb[2][2].v[k]
                 - nb[0][0].v[k] - 2.f * nb[1][0].v[k] - nb[2][0].v[k]) * 0.125f;
      float dyv = (nb[2][0].v[k] + 2.f * nb[2][1].v[k] + nb[2][2].v[k]
                 - nb[0][0].v[k] - 2.f * nb[0][1].v[k] - nb[0][2].v[k]) * 0.125f;
      dst[3 * k + 0] = (_Float16)id;
      dst[3 * k + 1] = (_Float16)dxv;
      dst[3 * k + 2] = (_Float16)dyv;
    }
  }
  { // zero-pad features 48..63
    _Float16* zp = yrow + 48 + (lo ? 0 : 8);
#pragma unroll
    for (int t = 0; t < 8; ++t) zp[t] = (_Float16)0.0f;
  }
  __syncthreads();

  // ---- GEMM1: h[16x128] = relu(y[16x64] @ w0h^T), f16 WMMA, f32 acc ----
  v16h a0 = frag_stride16(&y_lds[wave][m * 64 + 0  + hi8]);
  v16h a1 = frag_stride16(&y_lds[wave][m * 64 + 32 + hi8]);
#pragma unroll
  for (int nt = 0; nt < 8; ++nt) {
    const _Float16* bb = &w0h[(nt * 16 + m) * 64 + hi8];
    v16h b0 = frag_stride16(bb);
    v16h b1 = frag_stride16(bb + 32);
    v8f acc = {};
    acc = __builtin_amdgcn_wmma_f32_16x16x32_f16(false, a0, false, b0, (short)0, acc, false, false);
    acc = __builtin_amdgcn_wmma_f32_16x16x32_f16(false, a1, false, b1, (short)0, acc, false, false);
    // relu + scatter C/D layout (row = r + hi8, col = m) into A layout of GEMM2
    _Float16* hdst = &h_lds[wave][hi8 * 128 + nt * 16 + m];
#pragma unroll
    for (int r = 0; r < 8; ++r) {
      float v = acc[r];
      hdst[r * 128] = (_Float16)(v > 0.0f ? v : 0.0f);
    }
  }
  __syncthreads();

  // ---- GEMM2: dx[16x16] = h[16x128] @ w1t^T ----
  v8f acc2 = {};
#pragma unroll
  for (int kc = 0; kc < 4; ++kc) {
    v16h a  = frag_stride16(&h_lds[wave][m * 128 + kc * 32 + hi8]);
    v16h bb = frag_stride16(&w1t[m * 128 + kc * 32 + hi8]);
    acc2 = __builtin_amdgcn_wmma_f32_16x16x32_f16(false, a, false, bb, (short)0, acc2, false, false);
  }

  // ---- residual update; stash updated alpha for the life pass ----
  const size_t rowoff = ((size_t)b * Hn + hrow) * Wn;
  float rv = rmask[rowoff + wbase + m];     // rand for pixel (lane&15)
  const size_t pbase = (rowoff + wbase) * Cn;
#pragma unroll
  for (int r = 0; r < 8; ++r) {
    int pix = r + hi8;
    float rp  = __shfl(rv, pix, 32);
    float upd = (rp <= 0.5f) ? 1.0f : 0.0f; // FIRE_RATE
    size_t gi = pbase + (size_t)pix * Cn + m;
    float x2 = x[gi] + acc2[r] * upd;        // STEP_SIZE == 1
    out[gi] = x2;
    if (m == 3) alpha2[rowoff + wbase + pix] = x2;
  }
}

// Pass 2: life = (3x3 maxpool(x alpha) > thr) & (3x3 maxpool(x2 alpha) > thr);
// scale own pixel of `out` in place (no cross-block hazard).
__global__ __launch_bounds__(256) void life_mask_kernel(
    const float* __restrict__ x, const float* __restrict__ alpha2,
    float* __restrict__ out) {
  int p = blockIdx.x * 256 + threadIdx.x;
  int w    = p & (Wn - 1);
  int t    = p >> 7;            // / Wn
  int hrow = t & (Hn - 1);
  int b    = t >> 7;            // / Hn
  float pre = -1e30f, post = -1e30f;
#pragma unroll
  for (int i = -1; i <= 1; ++i) {
    int rr = hrow + i;
    if (rr < 0 || rr >= Hn) continue;
#pragma unroll
    for (int j = -1; j <= 1; ++j) {
      int qq = w + j;
      if (qq < 0 || qq >= Wn) continue;
      size_t q = ((size_t)b * Hn + rr) * Wn + qq;
      pre  = fmaxf(pre,  x[q * Cn + 3]);
      post = fmaxf(post, alpha2[q]);
    }
  }
  float life = (pre > 0.1f && post > 0.1f) ? 1.0f : 0.0f;
  size_t base = (size_t)p * Cn;
#pragma unroll
  for (int c = 0; c < 16; c += 4) {
    f4 v = *(f4*)(out + base + c);
    v.v[0] *= life; v.v[1] *= life; v.v[2] *= life; v.v[3] *= life;
    *(f4*)(out + base + c) = v;
  }
}

extern "C" void kernel_launch(void* const* d_in, const int* in_sizes, int n_in,
                              void* d_out, int out_size, void* d_ws, size_t ws_size,
                              hipStream_t stream) {
  (void)in_sizes; (void)n_in; (void)out_size; (void)ws_size;
  const float* x  = (const float*)d_in[0];
  const float* w0 = (const float*)d_in[1];
  const float* w1 = (const float*)d_in[2];
  const float* rm = (const float*)d_in[3];
  float* out    = (float*)d_out;
  float* alpha2 = (float*)d_ws;   // B*H*W floats = 2 MB scratch

  // 2 blocks per image row (64 px each), 4 waves * 16 px per block
  nca_step_kernel<<<dim3(Bn * Hn * 2), dim3(128), 0, stream>>>(x, w0, w1, rm, out, alpha2);
  life_mask_kernel<<<dim3((Bn * Hn * Wn) / 256), dim3(256), 0, stream>>>(x, alpha2, out);
}
